// FourierLinear_70772471103840
// MI455X (gfx1250) — matrix operand order repaired and checked by reading, not compiled, not run on hardware
//
#include <hip/hip_runtime.h>
#include <hip/hip_bf16.h>

// ---------------------------------------------------------------------------
// FourierLinear on MI455X (gfx1250):
//   Phase 0: x (f32) -> bf16, pre-swizzled into WMMA A-fragment layout (ws)
//   Phase 1: synthesize W (512x512) -> bf16 W^T (ws) + bias (ws)
//   Phase 2: out = x @ W + b via V_WMMA_F32_16X16X32_BF16, f32 accumulate.
//     Block = 8 waves sharing one 64-col N-strip; the W^T panel (64x512 bf16,
//     64KB + bank padding) is DMA'd into LDS once per block by the Tensor
//     Data Mover (tensor_load_to_lds, TENSORcnt), cutting global B traffic 8x.
//     Per K-step each wave does 2 coalesced A loads (global, pre-swizzled) +
//     8 conflict-free ds_load_b128 B reads + 8 WMMAs.
// ---------------------------------------------------------------------------

#define IN_DIM   512
#define OUT_DIM  512
#define TK       192              // TOTAL_K = 64 * 3 bands
#define M_ROWS   8192             // 8 * 1024
#define KSTEPS   (IN_DIM / 32)    // 16
#define LDS_ROW  520              // 512 elems + 8 pad elems (1040B, 260 DW == 4 mod 64 banks)
#define LDS_BYTES (64 * LDS_ROW * 2)   // 66560 B panel (<< 320KB WGP LDS)

typedef __attribute__((ext_vector_type(16))) __bf16         v16bf;
typedef __attribute__((ext_vector_type(8)))  float          v8f;
typedef __attribute__((ext_vector_type(16))) unsigned short v16u;
typedef __attribute__((ext_vector_type(4)))  unsigned int   u32x4;
typedef __attribute__((ext_vector_type(8)))  int            i32x8;
typedef __attribute__((ext_vector_type(4)))  int            i32x4;

#if __has_include(<hip/amd_detail/amd_gfx1250_TDM.h>)
#define TDM_6ARG 1
#else
#define TDM_6ARG 0
#endif

__device__ __forceinline__ unsigned short f32_to_bf16(float f) {
  unsigned u = __float_as_uint(f);
  u += 0x7FFFu + ((u >> 16) & 1u);      // round-to-nearest-even
  return (unsigned short)(u >> 16);
}

// ---------------------------------------------------------------------------
// TDM: DMA a 64-row x 512-col bf16 panel of W^T (row stride 512 elems in
// memory) into LDS at lds_off, padding +16B after each 1024B row so the LDS
// row stride is 1040B (bank-conflict-free b128 reads).  D# per ISA ch.8.
// ---------------------------------------------------------------------------
__device__ __forceinline__ void tdm_load_panel(const unsigned short* gptr,
                                               unsigned lds_off) {
  unsigned long long ga = (unsigned long long)(uintptr_t)gptr;
  u32x4 g0;
  g0[0] = 1u;                                        // count=1, user mode
  g0[1] = lds_off;                                   // lds_addr (bytes)
  g0[2] = (unsigned)(ga & 0xFFFFFFFFu);              // global_addr[31:0]
  g0[3] = (unsigned)((ga >> 32) & 0x01FFFFFFu)       // global_addr[56:32]
        | (2u << 30);                                // type=2 ("image")
  i32x8 g1;
  g1[0] = (1 << 16)          // data_size = 1 -> 2 bytes
        | (1 << 20)          // pad_enable
        | (7 << 22)          // pad_interval = 256 DWORDs (one 1024B row)
        | (3 << 25);         // pad_amount = 4 DWORDs (16B)
  g1[1] = (int)(512u << 16); // tensor_dim0[15:0] = 512 (atomic_barrier_addr=0)
  g1[2] = (int)(512u << 16); // tensor_dim0[31:16]=0 | tensor_dim1[15:0] = 512
  g1[3] = (int)(512u << 16); // tensor_dim1[31:16]=0 | tile_dim0 = 512
  g1[4] = 64;                // tile_dim1 = 64, tile_dim2 = 0 (2D)
  g1[5] = 512;               // tensor_dim0_stride[31:0] = 512 elems
  g1[6] = 0;                 // d0_stride[47:32]=0 | d1_stride[15:0]=0
  g1[7] = 0;
  i32x4 z4 = {0, 0, 0, 0};
#if TDM_6ARG
  i32x8 z8 = {0, 0, 0, 0, 0, 0, 0, 0};
  __builtin_amdgcn_tensor_load_to_lds(g0, g1, z4, z4, z8, 0);
#else
  __builtin_amdgcn_tensor_load_to_lds(g0, g1, z4, z4, 0);
#endif
}

// ---------------------------------------------------------------------------
// Phase 0: convert x to bf16 in A-fragment order (contiguous k: element s of lane half hi holds k = 16*hi + s, same as B).
// ---------------------------------------------------------------------------
__global__ void convert_x_kernel(const float* __restrict__ x,
                                 unsigned short* __restrict__ ax) {
  const int idx  = blockIdx.x * blockDim.x + threadIdx.x;  // 0..262143
  const int lane = idx & 31;
  const int ks   = (idx >> 5) & (KSTEPS - 1);
  const int mt   = idx >> 9;                               // 0..511
  const int ml   = lane & 15;
  const int hi   = lane >> 4;

  const float* xr = x + (size_t)(mt * 16 + ml) * IN_DIM + ks * 32 + hi * 16;
  v8f a0 = *(const v8f*)(xr);
  v8f a1 = *(const v8f*)(xr + 8);
  v16u au;
#pragma unroll
  for (int e = 0; e < 8; ++e) {
    au[e]     = f32_to_bf16(a0[e]);
    au[e + 8] = f32_to_bf16(a1[e]);
  }
  *(v16u*)(ax + (size_t)idx * 16) = au;   // 32B/lane, fully coalesced
}

// ---------------------------------------------------------------------------
// Phase 1a: synthesize W^T (bf16): wt[j*512+i] = W[i][j]
// ---------------------------------------------------------------------------
__global__ void synth_w_kernel(const float* __restrict__ omega,
                               const float* __restrict__ phi,
                               const float* __restrict__ alpha,
                               const int* __restrict__ layer_idx,
                               const int* __restrict__ num_layers,
                               unsigned short* __restrict__ wt) {
  __shared__ float sh[576 + 192 + 384];
  float* sho = sh;
  float* shp = sh + 576;
  float* sha = sh + 768;
  for (int t = threadIdx.x; t < 1152; t += blockDim.x) {
    float v;
    if (t < 576)      v = omega[t];
    else if (t < 768) v = phi[t - 576];
    else              v = alpha[t - 768];
    sh[t] = v;
  }
  __syncthreads();

  const int idx = blockIdx.x * blockDim.x + threadIdx.x;
  const int i = idx >> 9;
  const int j = idx & 511;

  const float lv = ((float)(*layer_idx) + 1.0f) / ((float)(*num_layers) + 1.0f);
  const float ii = (i + 1.0f) / (IN_DIM + 1.0f);
  const float jj = (j + 1.0f) / (OUT_DIM + 1.0f);

  float acc = 0.0f;
#pragma unroll 4
  for (int k = 0; k < TK; ++k) {
    float arg = fmaf(ii, sho[3 * k],
                fmaf(jj, sho[3 * k + 1],
                fmaf(lv, sho[3 * k + 2], shp[k])));
    float s, c;
    __sincosf(arg, &s, &c);
    acc = fmaf(sha[k], s, acc);
    acc = fmaf(sha[TK + k], c, acc);
  }
  const float XAVIER = 0.04419417382415922f;  // sqrt(2/(512+512))
  wt[j * IN_DIM + i] = f32_to_bf16(acc * XAVIER);
}

// ---------------------------------------------------------------------------
// Phase 1b: bias b[j] = 0.02 * sum_k (ab[k] sin + ab[192+k] cos)
// ---------------------------------------------------------------------------
__global__ void synth_b_kernel(const float* __restrict__ omega,
                               const float* __restrict__ phi,
                               const float* __restrict__ alpha_bias,
                               const int* __restrict__ layer_idx,
                               const int* __restrict__ num_layers,
                               float* __restrict__ bias) {
  const int j = blockIdx.x * blockDim.x + threadIdx.x;
  if (j >= OUT_DIM) return;
  const float lv = ((float)(*layer_idx) + 1.0f) / ((float)(*num_layers) + 1.0f);
  const float jj = (j + 1.0f) / (OUT_DIM + 1.0f);
  float acc = 0.0f;
#pragma unroll 4
  for (int k = 0; k < TK; ++k) {
    float arg = fmaf(jj, omega[3 * k + 1], fmaf(lv, omega[3 * k + 2], phi[k]));
    float s, c;
    __sincosf(arg, &s, &c);
    acc = fmaf(alpha_bias[k], s, acc);
    acc = fmaf(alpha_bias[TK + k], c, acc);
  }
  bias[j] = acc * 0.1f * 0.2f;   // SCALE * 0.2
}

// ---------------------------------------------------------------------------
// Phase 2 GEMM.  Grid (32, 8): blockIdx.x = 256-row M block, blockIdx.y =
// 64-col N strip.  8 waves/block, wave w -> rows [m0, m0+32).
// B panel comes from LDS (TDM-staged); A fragments from global (L2-resident).
// ---------------------------------------------------------------------------
__global__ void gemm_wmma_kernel(const unsigned short* __restrict__ ax,
                                 const unsigned short* __restrict__ wt,
                                 const float* __restrict__ bias,
                                 float* __restrict__ out) {
  extern __shared__ unsigned short bsm[];   // 64 rows x LDS_ROW bf16

  const int lane = threadIdx.x & 31;
  const int wave = threadIdx.x >> 5;
  const int n0   = blockIdx.y * 64;
  const int m0   = blockIdx.x * 256 + wave * 32;
  const int ml   = lane & 15;
  const int hi   = lane >> 4;

  // --- TDM: one wave DMAs the 64x512 bf16 W^T panel into LDS
  if (wave == 0) {
    tdm_load_panel(wt + (size_t)n0 * IN_DIM,
                   __builtin_amdgcn_groupstaticsize());
    __builtin_amdgcn_s_wait_tensorcnt(0);
  }
  __syncthreads();

  v8f c[8];
#pragma unroll
  for (int t = 0; t < 8; ++t) c[t] = (v8f){};

  const int mt2 = m0 >> 4;   // first 16-row tile index
  const unsigned short* ap0 = ax + ((size_t)mt2 * 16 * 32 + lane) * 16;
  const unsigned short* ap1 = ax + ((size_t)(mt2 + 1) * 16 * 32 + lane) * 16;
  // per-lane LDS B base: row = t*16 + ml, elems ks*32 + hi*16 + {0..15}
  const unsigned short* bs = bsm + (size_t)ml * LDS_ROW + hi * 16;

  for (int ks = 0; ks < KSTEPS; ++ks) {
    // --- A: two 32B coalesced fragment loads (zero VALU), prefetch ahead
    v16bf A0 = __builtin_bit_cast(v16bf, *(const v16u*)(ap0 + (size_t)ks * 512));
    v16bf A1 = __builtin_bit_cast(v16bf, *(const v16u*)(ap1 + (size_t)ks * 512));
    __builtin_prefetch(ap0 + (size_t)(ks + 2) * 512, 0, 3);

    // --- B: 4 tiles from LDS, 32B per lane, bank-conflict-free
    const unsigned short* bk = bs + ks * 32;
    v16bf B0 = __builtin_bit_cast(v16bf, *(const v16u*)(bk));
    v16bf B1 = __builtin_bit_cast(v16bf, *(const v16u*)(bk + 16 * LDS_ROW));
    v16bf B2 = __builtin_bit_cast(v16bf, *(const v16u*)(bk + 32 * LDS_ROW));
    v16bf B3 = __builtin_bit_cast(v16bf, *(const v16u*)(bk + 48 * LDS_ROW));

    c[0] = __builtin_amdgcn_wmma_f32_16x16x32_bf16(false, A0, false, B0,
                                                   (short)0, c[0], false, false);
    c[4] = __builtin_amdgcn_wmma_f32_16x16x32_bf16(false, A1, false, B0,
                                                   (short)0, c[4], false, false);
    c[1] = __builtin_amdgcn_wmma_f32_16x16x32_bf16(false, A0, false, B1,
                                                   (short)0, c[1], false, false);
    c[5] = __builtin_amdgcn_wmma_f32_16x16x32_bf16(false, A1, false, B1,
                                                   (short)0, c[5], false, false);
    c[2] = __builtin_amdgcn_wmma_f32_16x16x32_bf16(false, A0, false, B2,
                                                   (short)0, c[2], false, false);
    c[6] = __builtin_amdgcn_wmma_f32_16x16x32_bf16(false, A1, false, B2,
                                                   (short)0, c[6], false, false);
    c[3] = __builtin_amdgcn_wmma_f32_16x16x32_bf16(false, A0, false, B3,
                                                   (short)0, c[3], false, false);
    c[7] = __builtin_amdgcn_wmma_f32_16x16x32_bf16(false, A1, false, B3,
                                                   (short)0, c[7], false, false);
  }

  // --- epilogue: add bias, store per C/D layout (M = r + 8*hi, N = lane&15)
#pragma unroll
  for (int t = 0; t < 4; ++t) {
    const int n = n0 + t * 16 + ml;
    const float bv = bias[n];
#pragma unroll
    for (int a = 0; a < 2; ++a) {
      float* orow = out + (size_t)(m0 + a * 16 + hi * 8) * OUT_DIM + n;
      const v8f cc = c[a * 4 + t];
#pragma unroll
      for (int r = 0; r < 8; ++r) {
        orow[(size_t)r * OUT_DIM] = cc[r] + bv;
      }
    }
  }
}

extern "C" void kernel_launch(void* const* d_in, const int* in_sizes, int n_in,
                              void* d_out, int out_size, void* d_ws, size_t ws_size,
                              hipStream_t stream) {
  const float* x          = (const float*)d_in[0];
  const float* omega      = (const float*)d_in[1];
  const float* phi        = (const float*)d_in[2];
  const float* alpha      = (const float*)d_in[3];
  const float* alpha_bias = (const float*)d_in[4];
  const int*   layer_idx  = (const int*)d_in[5];
  const int*   num_layers = (const int*)d_in[6];
  float* out = (float*)d_out;

  // workspace: bf16 W^T (512KB) | f32 bias (2KB) | bf16 A-swizzled x (8MB)
  unsigned short* wt = (unsigned short*)d_ws;
  float* bias = (float*)((char*)d_ws + (size_t)IN_DIM * OUT_DIM * sizeof(unsigned short));
  unsigned short* axbuf = (unsigned short*)((char*)d_ws +
      (size_t)IN_DIM * OUT_DIM * sizeof(unsigned short) + OUT_DIM * sizeof(float));

  // Phase 0: x -> bf16 A fragments
  convert_x_kernel<<<(M_ROWS * IN_DIM / 16) / 256, 256, 0, stream>>>(x, axbuf);

  // Phase 1: synthesize W^T (bf16) and bias
  synth_w_kernel<<<(IN_DIM * OUT_DIM) / 256, 256, 0, stream>>>(
      omega, phi, alpha, layer_idx, num_layers, wt);
  synth_b_kernel<<<OUT_DIM / 256, 256, 0, stream>>>(
      omega, phi, alpha_bias, layer_idx, num_layers, bias);

  // Phase 2: WMMA GEMM + bias. Grid (32 M-blocks, 8 N-strips), 8 waves/block,
  // 66560B dynamic LDS for the TDM-staged B panel.
  gemm_wmma_kernel<<<dim3(32, 8), 256, LDS_BYTES, stream>>>(axbuf, wt, bias, out);
}
